// CausalSelfAttentionARMA_57389353009769
// MI455X (gfx1250) — matrix operand order repaired and checked by reading, not compiled, not run on hardware
//
#include <hip/hip_runtime.h>
#include <hip/hip_bf16.h>
#include <math.h>

// Problem constants (match reference)
#define BB    2
#define TT    2048
#define DD    1024
#define NHEAD 16
#define HD    64
#define CK    128   // key-chunk size in the attention kernels

typedef __attribute__((ext_vector_type(16))) _Float16 v16h;
typedef __attribute__((ext_vector_type(8)))  float    v8f;
typedef __attribute__((ext_vector_type(4)))  unsigned v4u;
typedef __attribute__((ext_vector_type(4)))  int      v4i;
typedef __attribute__((ext_vector_type(8)))  int      v8i;

__device__ __forceinline__ v8f wmma16(v16h a, v16h b, v8f c) {
  // D = A(16x32 f16) x B(32x16 f16) + C(16x16 f32)
  return __builtin_amdgcn_wmma_f32_16x16x32_f16(false, a, false, b, (short)0, c, false, false);
}

__device__ __forceinline__ v8f v8f_zero() {
  v8f z = {0.f, 0.f, 0.f, 0.f, 0.f, 0.f, 0.f, 0.f};
  return z;
}

// ---- Tensor Data Mover: stage a 2D f16 tile (rows x cols) into LDS -------
// D# packing per CDNA5 ISA section 8.3/8.4 (group0: count/lds/global/type,
// group1: data_size=2B, tensor dims (no clipping), tile dims, dim0 stride).
__device__ __forceinline__ void tdm_load_2d_f16(void* lds_dst, const _Float16* src,
                                                unsigned rows, unsigned cols,
                                                unsigned stride_elems) {
#if __has_builtin(__builtin_amdgcn_tensor_load_to_lds)
  unsigned long long ga = (unsigned long long)(uintptr_t)(const void*)src;
  unsigned lds_off = (unsigned)(uintptr_t)lds_dst;   // low 32 bits = LDS byte offset
  v4u g0;
  g0[0] = 1u;                                        // count=1, user descriptor
  g0[1] = lds_off;                                   // lds_addr (bytes)
  g0[2] = (unsigned)(ga & 0xffffffffu);              // global_addr[31:0]
  g0[3] = (unsigned)((ga >> 32) & 0x01ffffffu) | (2u << 30);  // ga[56:32] | type=2
  const unsigned td0 = 0x7fffffffu;                  // tensor_dim0 (huge: no OOB clip)
  const unsigned td1 = 0x7fffffffu;                  // tensor_dim1
  v8i g1;
  g1[0] = (int)(1u << 16);                           // data_size=1 (2 bytes/elem)
  g1[1] = (int)((td0 & 0xffffu) << 16);              // tensor_dim0 lo16 at bit 48
  g1[2] = (int)(((td0 >> 16) & 0xffffu) | ((td1 & 0xffffu) << 16));
  g1[3] = (int)(((td1 >> 16) & 0xffffu) | ((cols & 0xffffu) << 16));  // tile_dim0
  g1[4] = (int)(rows & 0xffffu);                     // tile_dim1 (tile_dim2=0)
  g1[5] = (int)stride_elems;                         // tensor_dim0_stride[31:0]
  g1[6] = 0;                                         // stride hi | dim1_stride lo
  g1[7] = 0;
  v4i g2 = {0, 0, 0, 0};
  v4i g3 = {0, 0, 0, 0};
#if __clang_major__ >= 23
  v8i g4 = {0, 0, 0, 0, 0, 0, 0, 0};
  __builtin_amdgcn_tensor_load_to_lds(g0, g1, g2, g3, g4, 0);
#else
  __builtin_amdgcn_tensor_load_to_lds(g0, g1, g2, g3, 0);
#endif
#endif
}

// Stage a tile: TDM path (wave 0 issues, async) or cooperative fallback.
__device__ __forceinline__ void stage2d(_Float16* lds, const _Float16* src,
                                        int rows, int cols, int stride_elems,
                                        int tid, int nthr) {
#if __has_builtin(__builtin_amdgcn_tensor_load_to_lds)
  if (tid < 32) tdm_load_2d_f16(lds, src, (unsigned)rows, (unsigned)cols,
                                (unsigned)stride_elems);
#else
  for (int i = tid; i < rows * cols; i += nthr)
    lds[i] = src[(size_t)(i / cols) * (size_t)stride_elems + (i % cols)];
#endif
}

// Wait for outstanding TDM transfers (issuing wave), then block barrier.
__device__ __forceinline__ void stage_wait(int tid) {
#if __has_builtin(__builtin_amdgcn_tensor_load_to_lds) && __has_builtin(__builtin_amdgcn_s_wait_tensorcnt)
  if (tid < 32) __builtin_amdgcn_s_wait_tensorcnt(0);
#endif
  __syncthreads();
}

// ---- WMMA fragment loaders (ISA 7.12.2 layouts) --------------------------
// A fragment 16x32: lane<16 -> row M=lane, K pairs {0..7} in v0..3, {16..23} in v4..7
//                   lane>=16 -> row M=lane-16, K pairs {8..15}, {24..31}
__device__ __forceinline__ v16h load_a_frag(const _Float16* base, int ld, int m0, int k0, int lane) {
  const int half = (lane >> 4) & 1;
  const int r    = lane & 15;
  const _Float16* p = base + (size_t)(m0 + r) * (size_t)ld + (size_t)(k0 + half * 8);
  union { v16h v; unsigned u[8]; } f;
#pragma unroll
  for (int j = 0; j < 4; ++j) f.u[j]     = *(const unsigned*)(p + 2 * j);
#pragma unroll
  for (int j = 0; j < 4; ++j) f.u[4 + j] = *(const unsigned*)(p + 16 + 2 * j);
  return f.v;
}

// B fragment 32x16 where B[k][n] = src[(n0+n)*ld + k0 + k]  (src is [N][K] row-major)
__device__ __forceinline__ v16h load_b_frag(const _Float16* base, int ld, int n0, int k0, int lane) {
  const int half = (lane >> 4) & 1;
  const int n    = lane & 15;
  const _Float16* p = base + (size_t)(n0 + n) * (size_t)ld + (size_t)(k0 + half * 8);
  union { v16h v; unsigned u[8]; } f;
#pragma unroll
  for (int j = 0; j < 4; ++j) f.u[j]     = *(const unsigned*)(p + 2 * j);
#pragma unroll
  for (int j = 0; j < 4; ++j) f.u[4 + j] = *(const unsigned*)(p + 16 + 2 * j);
  return f.v;
}

// half-wave (16-lane) reductions: rows of a C-fragment live across lanes 0..15 / 16..31
__device__ __forceinline__ float hw_max(float v) {
#pragma unroll
  for (int off = 1; off < 16; off <<= 1) v = fmaxf(v, __shfl_xor(v, off, 32));
  return v;
}
__device__ __forceinline__ float hw_sum(float v) {
#pragma unroll
  for (int off = 1; off < 16; off <<= 1) v += __shfl_xor(v, off, 32);
  return v;
}

// ---- elementwise prep kernels --------------------------------------------
__global__ void conv_f16_kernel(const float* __restrict__ in, _Float16* __restrict__ out, size_t n) {
  size_t i = (size_t)blockIdx.x * blockDim.x + threadIdx.x;
  if (i < n) out[i] = (_Float16)in[i];
}

__global__ void qaka_kernel(const _Float16* __restrict__ qkh, const _Float16* __restrict__ k2h,
                            _Float16* __restrict__ qah, _Float16* __restrict__ kah, size_t n) {
  size_t i = (size_t)blockIdx.x * blockDim.x + threadIdx.x;
  if (i >= n) return;
  size_t row = i >> 10, col = i & 1023;
  float q  = (float)qkh[(row << 11) + col];        // q part of the fused qk buffer
  float qa = (q < 0.f) ? q * 0.125f : q * (0.125f * 0.02f);
  qah[i] = (_Float16)qa;
  float k2 = (float)k2h[i];
  float z  = k2 * (0.125f * 0.02f);
  kah[i] = (_Float16)(1.f / (1.f + __expf(-z)));
}

// vT[b,h,d,t] = x[b,t,h*64+d]
__global__ void vT_kernel(const float* __restrict__ x, _Float16* __restrict__ vT) {
  size_t i = (size_t)blockIdx.x * blockDim.x + threadIdx.x;
  if (i >= (size_t)BB * NHEAD * HD * TT) return;
  int t = (int)(i % TT);
  int d = (int)((i / TT) % HD);
  int h = (int)((i / ((size_t)TT * HD)) % NHEAD);
  int b = (int)(i / ((size_t)TT * HD * NHEAD));
  vT[i] = (_Float16)x[((size_t)b * TT + t) * DD + h * HD + d];
}

// eT[b,h,d,t'] = x[b,t'+1,h*64+d] - y_ar[b,t',h*64+d]   (0 for t'=T-1)
__global__ void eT_kernel(const float* __restrict__ x, const float* __restrict__ yar,
                          _Float16* __restrict__ eT) {
  size_t i = (size_t)blockIdx.x * blockDim.x + threadIdx.x;
  if (i >= (size_t)BB * NHEAD * HD * TT) return;
  int t = (int)(i % TT);
  int d = (int)((i / TT) % HD);
  int h = (int)((i / ((size_t)TT * HD)) % NHEAD);
  int b = (int)(i / ((size_t)TT * HD * NHEAD));
  float v = 0.f;
  if (t < TT - 1) {
    size_t c = (size_t)h * HD + d;
    v = x[((size_t)b * TT + t + 1) * DD + c] - yar[((size_t)b * TT + t) * DD + c];
  }
  eT[i] = (_Float16)v;
}

__global__ void combine_kernel(const float* __restrict__ a, const float* __restrict__ b,
                               _Float16* __restrict__ o, size_t n) {
  size_t i = (size_t)blockIdx.x * blockDim.x + threadIdx.x;
  if (i < n) o[i] = (_Float16)(a[i] + b[i]);
}

// ---- TDM-staged, double-buffered WMMA GEMM -------------------------------
// C[M,N] = A[M,K] @ W[N,K]^T + bscale*bias[n].
// 128 threads (4 waves) per block compute a 64x128 tile. The TDM stages the
// NEXT k-panel into the alternate LDS buffer while WMMAs consume the current
// one; s_wait_tensorcnt + barrier only at the end of each k-step.
#define GBM 64
#define GBN 128
#define GBK 64
__global__ void gemm_wmma(const _Float16* __restrict__ A, int lda,
                          const _Float16* __restrict__ W, int ldw,
                          const float* __restrict__ bias, float bscale,
                          _Float16* __restrict__ Cf16, float* __restrict__ Cf32,
                          int ldc, int K) {
  __shared__ _Float16 Atile[2][GBM * GBK];   // 2 x 8 KB
  __shared__ _Float16 Btile[2][GBN * GBK];   // 2 x 16 KB
  const int tid  = threadIdx.x;
  const int lane = tid & 31;
  const int wave = tid >> 5;
  const int m0 = blockIdx.x * GBM;
  const int n0 = blockIdx.y * GBN;

  v8f acc[8];
#pragma unroll
  for (int d = 0; d < 8; ++d) acc[d] = v8f_zero();

  const int nsteps = K / GBK;
  // prologue: stage panel 0 and wait
  stage2d(Atile[0], A + (size_t)m0 * lda, GBM, GBK, lda, tid, 128);
  stage2d(Btile[0], W + (size_t)n0 * ldw, GBN, GBK, ldw, tid, 128);
  stage_wait(tid);

  for (int i = 0; i < nsteps; ++i) {
    const int cur = i & 1;
    const int nxt = cur ^ 1;
    if (i + 1 < nsteps) {           // issue next panel's TDM, no wait yet
      const int k = (i + 1) * GBK;
      stage2d(Atile[nxt], A + (size_t)m0 * lda + k, GBM, GBK, lda, tid, 128);
      stage2d(Btile[nxt], W + (size_t)n0 * ldw + k, GBN, GBK, ldw, tid, 128);
    }
#pragma unroll
    for (int kk = 0; kk < GBK; kk += 32) {
      v16h a = load_a_frag(Atile[cur], GBK, wave * 16, kk, lane);
#pragma unroll
      for (int d = 0; d < 8; ++d) {
        v16h b = load_b_frag(Btile[cur], GBK, d * 16, kk, lane);
        acc[d] = wmma16(a, b, acc[d]);
      }
    }
    if (i + 1 < nsteps) stage_wait(tid);   // transfer done + all waves flip
  }

  const int half = lane >> 4, nc = lane & 15;
#pragma unroll
  for (int d = 0; d < 8; ++d) {
    const int n = n0 + d * 16 + nc;
    const float bv = bias[n] * bscale;
#pragma unroll
    for (int r = 0; r < 8; ++r) {
      int m = m0 + wave * 16 + r + 8 * half;
      float v = acc[d][r] + bv;
      if (Cf16) Cf16[(size_t)m * ldc + n] = (_Float16)v;
      else      Cf32[(size_t)m * ldc + n] = v;
    }
  }
}

// ---- causal flash attention: 4 waves per block, 64 queries, shared K/V ---
__global__ void attn_flash(const _Float16* __restrict__ qk,   // [B*T][2048] (q | k)
                           const _Float16* __restrict__ vT,   // [B*H][64][T]
                           float* __restrict__ yar) {         // [B*T][1024]
  __shared__ _Float16 Ktile[CK * HD];          // 16 KB: K chunk  [key][dim]
  __shared__ _Float16 Vtile[HD * CK];          // 16 KB: V^T chunk [dim][key]
  __shared__ _Float16 Ptile[4][16 * CK];       // 16 KB: per-wave P
  const int tid  = threadIdx.x;
  const int lane = tid & 31;
  const int wave = tid >> 5;
  const int q0b = blockIdx.x * 64;
  const int q0  = q0b + wave * 16;
  const int h = blockIdx.y;
  const int b = blockIdx.z;
  const int ldq = 2 * DD;
  const _Float16* Qb = qk + (size_t)b * TT * ldq + h * HD;
  const _Float16* Kb = qk + (size_t)b * TT * ldq + DD + h * HD;
  const _Float16* Vt = vT + (size_t)(b * NHEAD + h) * HD * TT;

  const v16h aQ0 = load_a_frag(Qb, ldq, q0, 0, lane);
  const v16h aQ1 = load_a_frag(Qb, ldq, q0, 32, lane);

  float m[8], l[8];
  v8f o[4];
#pragma unroll
  for (int r = 0; r < 8; ++r) { m[r] = -__builtin_inff(); l[r] = 0.f; }
#pragma unroll
  for (int d = 0; d < 4; ++d) o[d] = v8f_zero();

  const int half = lane >> 4, nc = lane & 15;
  const int nchunk = (q0b + 63) / CK + 1;      // block-uniform
  const float scale = 0.125f;                  // 1/sqrt(64)

  for (int cidx = 0; cidx < nchunk; ++cidx) {
    const int c0 = cidx * CK;
    __syncthreads();                           // previous chunk's readers done
    stage2d(Ktile, Kb + (size_t)c0 * ldq, CK, HD, ldq, tid, 128);
    stage2d(Vtile, Vt + c0, HD, CK, TT, tid, 128);
    stage_wait(tid);

    v8f sf[8];
    float cm[8];
#pragma unroll
    for (int r = 0; r < 8; ++r) cm[r] = -__builtin_inff();
#pragma unroll
    for (int s = 0; s < 8; ++s) {
      v16h b0 = load_b_frag(Ktile, HD, s * 16, 0, lane);
      v16h b1 = load_b_frag(Ktile, HD, s * 16, 32, lane);
      v8f cc = wmma16(aQ1, b1, wmma16(aQ0, b0, v8f_zero()));
      const int kcol = c0 + s * 16 + nc;
#pragma unroll
      for (int r = 0; r < 8; ++r) {
        const int trow = q0 + r + 8 * half;
        float sv = (kcol <= trow) ? cc[r] * scale : -__builtin_inff();
        cc[r] = sv;
        cm[r] = fmaxf(cm[r], sv);
      }
      sf[s] = cc;
    }
    float alpha[8], rs[8];
#pragma unroll
    for (int r = 0; r < 8; ++r) {
      cm[r] = hw_max(cm[r]);
      float mn = fmaxf(m[r], cm[r]);
      alpha[r] = __expf(m[r] - mn);            // first chunk: exp(-inf-finite)=0
      m[r] = mn;
      rs[r] = 0.f;
    }
#pragma unroll
    for (int s = 0; s < 8; ++s) {
#pragma unroll
      for (int r = 0; r < 8; ++r) {
        float p = __expf(sf[s][r] - m[r]);     // masked -> 0
        rs[r] += p;
        Ptile[wave][(r + 8 * half) * CK + s * 16 + nc] = (_Float16)p;
      }
    }
#pragma unroll
    for (int r = 0; r < 8; ++r) {
      l[r] = l[r] * alpha[r] + hw_sum(rs[r]);
#pragma unroll
      for (int d = 0; d < 4; ++d) o[d][r] *= alpha[r];
    }
#pragma unroll
    for (int kk = 0; kk < 4; ++kk) {
      v16h aP = load_a_frag(Ptile[wave], CK, 0, kk * 32, lane);
#pragma unroll
      for (int d = 0; d < 4; ++d) {
        v16h bV = load_b_frag(Vtile, CK, d * 16, kk * 32, lane);
        o[d] = wmma16(aP, bV, o[d]);
      }
    }
  }
#pragma unroll
  for (int d = 0; d < 4; ++d)
#pragma unroll
    for (int r = 0; r < 8; ++r) {
      const int t = q0 + r + 8 * half;
      yar[((size_t)b * TT + t) * DD + h * HD + d * 16 + nc] = o[d][r] / l[r];
    }
}

// ---- causal linear attention: y_ma[t] = sum_{j<t} (qa[t].ka[j]) * e[j] ---
__global__ void ma_flash(const _Float16* __restrict__ qa,  // [B*T][1024]
                         const _Float16* __restrict__ ka,  // [B*T][1024]
                         const _Float16* __restrict__ eT,  // [B*H][64][T]
                         float* __restrict__ yma) {        // [B*T][1024]
  __shared__ _Float16 Katile[CK * HD];
  __shared__ _Float16 Etile[HD * CK];
  __shared__ _Float16 Ptile[4][16 * CK];
  const int tid  = threadIdx.x;
  const int lane = tid & 31;
  const int wave = tid >> 5;
  const int q0b = blockIdx.x * 64;
  const int q0  = q0b + wave * 16;
  const int h = blockIdx.y;
  const int b = blockIdx.z;
  const _Float16* Qb = qa + (size_t)b * TT * DD + h * HD;
  const _Float16* Kb = ka + (size_t)b * TT * DD + h * HD;
  const _Float16* Et = eT + (size_t)(b * NHEAD + h) * HD * TT;

  const v16h aQ0 = load_a_frag(Qb, DD, q0, 0, lane);
  const v16h aQ1 = load_a_frag(Qb, DD, q0, 32, lane);

  v8f o[4];
#pragma unroll
  for (int d = 0; d < 4; ++d) o[d] = v8f_zero();

  const int half = lane >> 4, nc = lane & 15;
  const int nchunk = (q0b + 62) / CK + 1;      // keys go up to (q0b+63)-1

  for (int cidx = 0; cidx < nchunk; ++cidx) {
    const int c0 = cidx * CK;
    __syncthreads();
    stage2d(Katile, Kb + (size_t)c0 * DD, CK, HD, DD, tid, 128);
    stage2d(Etile, Et + c0, HD, CK, TT, tid, 128);
    stage_wait(tid);
#pragma unroll
    for (int s = 0; s < 8; ++s) {
      v16h b0 = load_b_frag(Katile, HD, s * 16, 0, lane);
      v16h b1 = load_b_frag(Katile, HD, s * 16, 32, lane);
      v8f cc = wmma16(aQ1, b1, wmma16(aQ0, b0, v8f_zero()));
      const int kcol = c0 + s * 16 + nc;
#pragma unroll
      for (int r = 0; r < 8; ++r) {
        const int trow = q0 + r + 8 * half;
        float mv = (kcol < trow) ? cc[r] : 0.f; // strictly-causal, no softmax
        Ptile[wave][(r + 8 * half) * CK + s * 16 + nc] = (_Float16)mv;
      }
    }
#pragma unroll
    for (int kk = 0; kk < 4; ++kk) {
      v16h aP = load_a_frag(Ptile[wave], CK, 0, kk * 32, lane);
#pragma unroll
      for (int d = 0; d < 4; ++d) {
        v16h bE = load_b_frag(Etile, CK, d * 16, kk * 32, lane);
        o[d] = wmma16(aP, bE, o[d]);
      }
    }
  }
#pragma unroll
  for (int d = 0; d < 4; ++d)
#pragma unroll
    for (int r = 0; r < 8; ++r) {
      const int t = q0 + r + 8 * half;
      yma[((size_t)b * TT + t) * DD + h * HD + d * 16 + nc] = o[d][r];
    }
}

// ---- launch ---------------------------------------------------------------
extern "C" void kernel_launch(void* const* d_in, const int* in_sizes, int n_in,
                              void* d_out, int out_size, void* d_ws, size_t ws_size,
                              hipStream_t stream) {
  const float* x      = (const float*)d_in[0];
  const float* w_attn = (const float*)d_in[1];
  const float* b_attn = (const float*)d_in[2];
  const float* w_k2   = (const float*)d_in[3];
  const float* b_k2   = (const float*)d_in[4];
  const float* w_proj = (const float*)d_in[5];
  const float* b_proj = (const float*)d_in[6];
  float* out = (float*)d_out;

  const size_t BT = (size_t)BB * TT;
  char* ws = (char*)d_ws;
  size_t off = 0;
  auto alloc = [&](size_t bytes) -> char* {
    char* p = ws + off;
    off += (bytes + 255) & ~(size_t)255;
    return p;
  };

  _Float16* xh   = (_Float16*)alloc(BT * DD * 2);
  _Float16* wAh  = (_Float16*)alloc((size_t)2 * DD * DD * 2);
  _Float16* wK2h = (_Float16*)alloc((size_t)DD * DD * 2);
  _Float16* wPh  = (_Float16*)alloc((size_t)DD * DD * 2);
  _Float16* qkh  = (_Float16*)alloc(BT * 2 * DD * 2);
  _Float16* k2h  = (_Float16*)alloc(BT * DD * 2);
  _Float16* qah  = (_Float16*)alloc(BT * DD * 2);
  _Float16* kah  = (_Float16*)alloc(BT * DD * 2);
  _Float16* vT   = (_Float16*)alloc(BT * DD * 2);
  _Float16* eT   = (_Float16*)alloc(BT * DD * 2);
  float*    yar  = (float*)alloc(BT * DD * 4);
  float*    yma  = (float*)alloc(BT * DD * 4);
  _Float16* ysum = (_Float16*)alloc(BT * DD * 2);

  const int TB = 256;
  auto grid1 = [&](size_t n) { return dim3((unsigned)((n + TB - 1) / TB)); };

  // 1) fp32 -> fp16 conversions
  conv_f16_kernel<<<grid1(BT * DD), TB, 0, stream>>>(x, xh, BT * DD);
  conv_f16_kernel<<<grid1((size_t)2 * DD * DD), TB, 0, stream>>>(w_attn, wAh, (size_t)2 * DD * DD);
  conv_f16_kernel<<<grid1((size_t)DD * DD), TB, 0, stream>>>(w_k2, wK2h, (size_t)DD * DD);
  conv_f16_kernel<<<grid1((size_t)DD * DD), TB, 0, stream>>>(w_proj, wPh, (size_t)DD * DD);

  // 2) qk = x @ w_attn^T + b_attn      (M=4096, N=2048, K=1024)
  gemm_wmma<<<dim3(BT / GBM, (2 * DD) / GBN), 128, 0, stream>>>(
      xh, DD, wAh, DD, b_attn, 1.f, qkh, nullptr, 2 * DD, DD);
  // 3) k2 = x @ w_k2^T + b_k2          (M=4096, N=1024, K=1024)
  gemm_wmma<<<dim3(BT / GBM, DD / GBN), 128, 0, stream>>>(
      xh, DD, wK2h, DD, b_k2, 1.f, k2h, nullptr, DD, DD);

  // 4) vT transpose
  vT_kernel<<<grid1(BT * DD), TB, 0, stream>>>(x, vT);

  // 5) causal softmax attention -> y_ar
  attn_flash<<<dim3(TT / 64, NHEAD, BB), 128, 0, stream>>>(qkh, vT, yar);

  // 6) eT = (v[t+1] - y_ar[t])^T
  eT_kernel<<<grid1(BT * DD), TB, 0, stream>>>(x, yar, eT);

  // 7) qa / ka activations
  qaka_kernel<<<grid1(BT * DD), TB, 0, stream>>>(qkh, k2h, qah, kah, BT * DD);

  // 8) causal linear attention -> y_ma
  ma_flash<<<dim3(TT / 64, NHEAD, BB), 128, 0, stream>>>(qah, kah, eT, yma);

  // 9) combine
  combine_kernel<<<grid1(BT * DD), TB, 0, stream>>>(yar, yma, ysum, BT * DD);

  // 10) out = (y_ar + y_ma) @ w_proj^T + 2*b_proj   (f32 output)
  gemm_wmma<<<dim3(BT / GBM, DD / GBN), 128, 0, stream>>>(
      ysum, DD, wPh, DD, b_proj, 2.f, nullptr, out, DD, DD);
}
